// ConvTransformer_80513456931528
// MI455X (gfx1250) — compile-verified, matrix operand order checked by reading, not run on hardware
//
#include <hip/hip_runtime.h>
#include <hip/hip_bf16.h>
#include <math.h>

// ---------------- problem constants ----------------
constexpr int kB    = 4;
constexpr int kN    = 4096;
constexpr int kFeat = 64;
constexpr int kHid  = 128;
constexpr int kK    = 16;      // neighbors
constexpr int kGH   = 256;
constexpr int kGW   = 256;
constexpr int kGP   = 258;     // GH+2
constexpr int kSPC  = 96;      // 66 channels zero-padded to 96 (3 x K=32 WMMA steps)
constexpr int kNT   = kN / 16; // 16-point tiles per batch

typedef _Float16 f16;
typedef __attribute__((ext_vector_type(16))) _Float16 v16h;
typedef __attribute__((ext_vector_type(8)))  _Float16 v8h;
typedef __attribute__((ext_vector_type(8)))  float    v8f;

// ---------------- WMMA fragment helpers (wave32, 16x16x32 f16 -> f32) -------
// A (16x32 MxK) from a row-major source: per lane the 16 halves are two
// contiguous 16-byte runs: [row*ld + k0 + 8*grp .. +7] and [.. +16 .. +23].
__device__ __forceinline__ v16h load_a(const f16* p, int ld, int k0) {
  const int lane = threadIdx.x & 31;
  const int row = lane & 15, grp = (lane >> 4) & 1;
  const f16* base = p + row * ld + k0 + 8 * grp;
  v8h lo = *(const v8h*)(base);       // K = k0+8g .. k0+8g+7
  v8h hi = *(const v8h*)(base + 16);  // K = k0+16+8g .. k0+23+8g
  return __builtin_shufflevector(lo, hi, 0, 1, 2, 3, 4, 5, 6, 7,
                                         8, 9, 10, 11, 12, 13, 14, 15);
}
// B fragments are pre-packed: fragment f, lane l at  pk[(f*32 + l)*16 .. +15]
__device__ __forceinline__ v16h load_pk(const f16* pk, int frag) {
  const int lane = threadIdx.x & 31;
  return *(const v16h*)(pk + (((size_t)frag << 5) + lane) * 16);
}
#define WMMA_F16(a, b, c) \
  __builtin_amdgcn_wmma_f32_16x16x32_f16(false, (a), false, (b), (short)0, (c), false, false)
// C/D layout: lane holds column (lane&15); acc[r] is row r + 8*(lane>>4).

// ---------------- weight pre-pack into fragment order ----------------
// Source: nmat matrices, each kvalid x Nd row-major f32 (K zero-padded to ksteps*32).
// Output fragment order: [nt][mat][ks][lane][16 halves].
__global__ void pack_b_kernel(const float* __restrict__ W, f16* __restrict__ out,
                              int nmat, int ksteps, int ntiles, int Nd, int kvalid) {
  int o = blockIdx.x * 256 + threadIdx.x;
  int total = nmat * ksteps * ntiles * 512;
  if (o >= total) return;
  int f = o >> 9;
  int lane = (o >> 4) & 31;
  int h = o & 15;
  int nt  = f / (nmat * ksteps);
  int rem = f % (nmat * ksteps);
  int mat = rem / ksteps;
  int ks  = rem % ksteps;
  int col = lane & 15, grp = lane >> 4;
  int v = h >> 1, j = h & 1;
  int kb = ks * 32 + 16 * (v >> 2) + 2 * (v & 3) + 8 * grp + j;
  float val = (kb < kvalid) ? W[((size_t)mat * kvalid + kb) * Nd + nt * 16 + col] : 0.f;
  out[o] = (f16)val;
}

// ---------------- small utility kernels ----------------
__global__ void fill_grid_kernel(int* __restrict__ g, int n) {
  int i = blockIdx.x * 256 + threadIdx.x;
  if (i < n) g[i] = -1;
}

__global__ void scatter_grid_kernel(const float* __restrict__ xyzp, int* __restrict__ g) {
  int i = blockIdx.x * 256 + threadIdx.x;
  if (i >= kB * kN) return;
  int b = i / kN, n = i % kN;
  int cx = __float2int_rn(xyzp[(size_t)i * 4 + 0] * kGW);
  int cy = __float2int_rn(xyzp[(size_t)i * 4 + 1] * kGH);
  g[(size_t)b * kGP * kGP + (size_t)(cy + 1) * kGP + (cx + 1)] = n;
}

// ---------------- KNN: one thread per query, all xy tiled in LDS ------------
__global__ void knn_kernel(const float* __restrict__ xyzp, int* __restrict__ knn_out) {
  __shared__ float sx[kN];
  __shared__ float sy[kN];
  const int tid = threadIdx.x;
  const int bpb = kN / 256;
  const int b = blockIdx.x / bpb;
  const int n0 = (blockIdx.x % bpb) * 256;
  for (int i = tid; i < kN; i += 256) {
    sx[i] = xyzp[(size_t)(b * kN + i) * 4 + 0];
    sy[i] = xyzp[(size_t)(b * kN + i) * 4 + 1];
  }
  __syncthreads();
  const int n = n0 + tid;
  const float qx = sx[n], qy = sy[n];
  float bd[kK];
  int   bi[kK];
#pragma unroll
  for (int j = 0; j < kK; ++j) { bd[j] = 3.4e38f; bi[j] = 0; }
  for (int m = 0; m < kN; ++m) {
    float dx = sx[m] - qx, dy = sy[m] - qy;
    float d = dx * dx + dy * dy;
    if (d < bd[kK - 1]) {          // strict < keeps earlier index on ties (stable)
      float cd = d; int ci = m;
#pragma unroll
      for (int j = 0; j < kK; ++j) {
        if (cd < bd[j]) {
          float td = bd[j]; int ti = bi[j];
          bd[j] = cd; bi[j] = ci; cd = td; ci = ti;
        }
      }
    }
  }
#pragma unroll
  for (int j = 0; j < kK; ++j) knn_out[(size_t)(b * kN + n) * kK + j] = bi[j];
}

// ---------------- sp + LayerNorm (66 ch), f16 out padded to 96 --------------
__global__ void spln_kernel(const float* __restrict__ xyzp, const float* __restrict__ feat,
                            const float* __restrict__ ln_g, const float* __restrict__ ln_b,
                            f16* __restrict__ sp16) {
  const int lane = threadIdx.x & 31;
  const int wave = (blockIdx.x * blockDim.x + threadIdx.x) >> 5;
  if (wave >= kB * kN) return;
  const size_t pt = (size_t)wave;
  float p = (xyzp[pt * 4 + 3] - 0.5f) * 2.0f;
  float v0 = fminf(fmaxf(p, 0.f), 1.f);
  float v1 = -fminf(fmaxf(p, -1.f), 0.f);
  auto chv = [&](int ch) -> float {
    if (ch == 0) return v0;
    if (ch == 1) return v1;
    return feat[pt * kFeat + (ch - 2)];
  };
  float a0 = chv(lane);
  float a1 = chv(lane + 32);
  float a2 = (lane < 2) ? chv(lane + 64) : 0.f;
  float s = a0 + a1 + a2, sq = a0 * a0 + a1 * a1 + a2 * a2;
#pragma unroll
  for (int off = 16; off; off >>= 1) {
    s  += __shfl_xor(s, off, 32);
    sq += __shfl_xor(sq, off, 32);
  }
  float mu  = s / 66.f;
  float var = sq / 66.f - mu * mu;
  float inv = rsqrtf(var + 1e-5f);
  size_t base = pt * kSPC;
  sp16[base + lane]      = (f16)((a0 - mu) * inv * ln_g[lane]      + ln_b[lane]);
  sp16[base + lane + 32] = (f16)((a1 - mu) * inv * ln_g[lane + 32] + ln_b[lane + 32]);
  if (lane < 2)
    sp16[base + lane + 64] = (f16)((a2 - mu) * inv * ln_g[lane + 64] + ln_b[lane + 64]);
  else
    sp16[base + lane + 64] = (f16)0.f;   // zero pad channels 66..95
}

// ---------------- conv stage 1: h1 = sum_tap (sp[nbr]*mask) @ W1 + b1 -------
// 128 threads = 4 waves; staging shared, n-tiles split across waves.
__global__ void __launch_bounds__(128) conv1_kernel(const float* __restrict__ xyzp,
    const int* __restrict__ grid, const f16* __restrict__ sp16,
    const f16* __restrict__ w1pk, const float* __restrict__ b1, f16* __restrict__ h1) {
  __shared__ alignas(16) f16 As[9][16][kSPC];
  __shared__ int nbr[16][9];
  const int tid = threadIdx.x;
  const int b  = blockIdx.x / kNT;
  const int n0 = (blockIdx.x % kNT) * 16;
  for (int e = tid; e < 16 * 9; e += 128) {
    int pt = e / 9, tap = e % 9;
    size_t n = (size_t)(b * kN + n0 + pt);
    int cx = __float2int_rn(xyzp[n * 4 + 0] * kGW);
    int cy = __float2int_rn(xyzp[n * 4 + 1] * kGH);
    int dy = tap / 3 - 1, dx = tap % 3 - 1;
    nbr[pt][tap] = grid[(size_t)b * kGP * kGP + (size_t)(cy + 1 + dy) * kGP + (cx + 1 + dx)];
  }
  __syncthreads();
  for (int e = tid; e < 9 * 16 * kSPC; e += 128) {
    int tap = e / (16 * kSPC);
    int rem = e % (16 * kSPC);
    int row = rem / kSPC, c = rem % kSPC;
    int id = nbr[row][tap];
    As[tap][row][c] = (id >= 0) ? sp16[(size_t)(b * kN + id) * kSPC + c] : (f16)0.f;
  }
  __syncthreads();
  const int wid = tid >> 5;
  const int lane = tid & 31;
  const int col = lane & 15, grp = (lane >> 4) & 1;
  for (int nt = wid; nt < 8; nt += 4) {
    v8f acc = {};
    for (int tap = 0; tap < 9; ++tap) {
#pragma unroll
      for (int ks = 0; ks < 3; ++ks)
        acc = WMMA_F16(load_a(&As[tap][0][0], kSPC, ks * 32),
                       load_pk(w1pk, (nt * 9 + tap) * 3 + ks), acc);
    }
    int j = nt * 16 + col;
    float bb = b1[j];
#pragma unroll
    for (int r = 0; r < 8; ++r)
      h1[(size_t)(b * kN + n0 + r + 8 * grp) * kHid + j] = (f16)(acc[r] + bb);
  }
}

// ---------------- conv stage 2: qkv = sum_tap (h1[nbr]*mask) @ W2 -----------
__global__ void __launch_bounds__(128) conv2_kernel(const float* __restrict__ xyzp,
    const int* __restrict__ grid, const f16* __restrict__ h1, const f16* __restrict__ w2pk,
    f16* __restrict__ q16, f16* __restrict__ k16, f16* __restrict__ v16) {
  __shared__ alignas(16) f16 As[9][16][kHid];
  __shared__ int nbr[16][9];
  const int tid = threadIdx.x;
  const int b  = blockIdx.x / kNT;
  const int n0 = (blockIdx.x % kNT) * 16;
  for (int e = tid; e < 16 * 9; e += 128) {
    int pt = e / 9, tap = e % 9;
    size_t n = (size_t)(b * kN + n0 + pt);
    int cx = __float2int_rn(xyzp[n * 4 + 0] * kGW);
    int cy = __float2int_rn(xyzp[n * 4 + 1] * kGH);
    int dy = tap / 3 - 1, dx = tap % 3 - 1;
    nbr[pt][tap] = grid[(size_t)b * kGP * kGP + (size_t)(cy + 1 + dy) * kGP + (cx + 1 + dx)];
  }
  __syncthreads();
  for (int e = tid; e < 9 * 16 * kHid; e += 128) {
    int tap = e / (16 * kHid);
    int rem = e % (16 * kHid);
    int row = rem / kHid, c = rem % kHid;
    int id = nbr[row][tap];
    As[tap][row][c] = (id >= 0) ? h1[(size_t)(b * kN + id) * kHid + c] : (f16)0.f;
  }
  __syncthreads();
  const int wid = tid >> 5;
  const int lane = tid & 31;
  const int col = lane & 15, grp = (lane >> 4) & 1;
  for (int nt = wid; nt < 24; nt += 4) {
    v8f acc = {};
    for (int tap = 0; tap < 9; ++tap) {
#pragma unroll
      for (int ks = 0; ks < 4; ++ks)
        acc = WMMA_F16(load_a(&As[tap][0][0], kHid, ks * 32),
                       load_pk(w2pk, (nt * 9 + tap) * 4 + ks), acc);
    }
    int j = nt * 16 + col;          // 0..383 -> (h = j/3, comp = j%3), reshape (HID,3)
    int h = j / 3, comp = j % 3;
    f16* dst = (comp == 0) ? q16 : (comp == 1) ? k16 : v16;
#pragma unroll
    for (int r = 0; r < 8; ++r)
      dst[(size_t)(b * kN + n0 + r + 8 * grp) * kHid + h] = (f16)acc[r];
  }
}

// ---------------- attention: one wave per point (16 nbrs x 128 feats) -------
__global__ void __launch_bounds__(32) attn_kernel(const float* __restrict__ xyzp,
    const int* __restrict__ knn_in,
    const f16* __restrict__ q16, const f16* __restrict__ k16, const f16* __restrict__ v16,
    const float* __restrict__ pe_w1, const float* __restrict__ pe_b1,
    const f16* __restrict__ pew2pk, const float* __restrict__ pe_b2,
    const f16* __restrict__ atw1pk, const float* __restrict__ at_b1,
    const f16* __restrict__ atw2pk, const float* __restrict__ at_b2,
    f16* __restrict__ res16) {
  __shared__ alignas(16) f16  A[16][kHid];
  __shared__ alignas(16) f16  T[16][kHid];
  __shared__ float peF[16][kHid];
  __shared__ int  sidx[16];
  const int lane = threadIdx.x;
  const int pid = blockIdx.x;
  const int b = pid / kN;
  if (lane < 16) sidx[lane] = knn_in[(size_t)pid * kK + lane];
  __syncthreads();
  const float sxv = xyzp[(size_t)pid * 4 + 0];
  const float syv = xyzp[(size_t)pid * 4 + 1];
  const float spv = xyzp[(size_t)pid * 4 + 3];
  // t1 = relu(rel @ pe_w1 + pe_b1)  (K=3 contraction: plain FMA, f32)
  for (int e = lane; e < 16 * kHid; e += 32) {
    int m = e >> 7, c = e & (kHid - 1);
    size_t nb = (size_t)(b * kN + sidx[m]);
    float rx = sxv - xyzp[nb * 4 + 0];
    float ry = syv - xyzp[nb * 4 + 1];
    float rp = spv - xyzp[nb * 4 + 3];
    float t = pe_b1[c] + rx * pe_w1[c] + ry * pe_w1[kHid + c] + rp * pe_w1[2 * kHid + c];
    A[m][c] = (f16)fmaxf(t, 0.f);
  }
  __syncthreads();
  const int col = lane & 15, grp = (lane >> 4) & 1;
  // pe = t1 @ pe_w2 + pe_b2 (kept f32 in LDS, reused twice)
  for (int nt = 0; nt < 8; ++nt) {
    v8f acc = {};
#pragma unroll
    for (int ks = 0; ks < 4; ++ks)
      acc = WMMA_F16(load_a(&A[0][0], kHid, ks * 32), load_pk(pew2pk, nt * 4 + ks), acc);
    int j = nt * 16 + col;
    float bb = pe_b2[j];
#pragma unroll
    for (int r = 0; r < 8; ++r) peF[r + 8 * grp][j] = acc[r] + bb;
  }
  __syncthreads();
  // a0 = q - k[idx] + pe
  for (int e = lane; e < 16 * kHid; e += 32) {
    int m = e >> 7, c = e & (kHid - 1);
    float qv = (float)q16[(size_t)pid * kHid + c];
    float kv = (float)k16[(size_t)(b * kN + sidx[m]) * kHid + c];
    A[m][c] = (f16)(qv - kv + peF[m][c]);
  }
  __syncthreads();
  // a1 = relu(a0 @ at_w1 + at_b1)
  for (int nt = 0; nt < 8; ++nt) {
    v8f acc = {};
#pragma unroll
    for (int ks = 0; ks < 4; ++ks)
      acc = WMMA_F16(load_a(&A[0][0], kHid, ks * 32), load_pk(atw1pk, nt * 4 + ks), acc);
    int j = nt * 16 + col;
    float bb = at_b1[j];
#pragma unroll
    for (int r = 0; r < 8; ++r) T[r + 8 * grp][j] = (f16)fmaxf(acc[r] + bb, 0.f);
  }
  __syncthreads();
  // a2 = a1 @ at_w2 + at_b2 ; softmax over the 16-neighbor axis ; weighted sum
  const float scale = 0.08838834764831845f; // 1/sqrt(128)
  for (int nt = 0; nt < 8; ++nt) {
    v8f acc = {};
#pragma unroll
    for (int ks = 0; ks < 4; ++ks)
      acc = WMMA_F16(load_a(&T[0][0], kHid, ks * 32), load_pk(atw2pk, nt * 4 + ks), acc);
    int j = nt * 16 + col;
    float bb = at_b2[j];
    float vals[8];
    float mx = -3.4e38f;
#pragma unroll
    for (int r = 0; r < 8; ++r) { vals[r] = (acc[r] + bb) * scale; mx = fmaxf(mx, vals[r]); }
    mx = fmaxf(mx, __shfl_xor(mx, 16, 32));          // column lives in lanes {j, j+16}
    float sum = 0.f;
#pragma unroll
    for (int r = 0; r < 8; ++r) { vals[r] = expf(vals[r] - mx); sum += vals[r]; }
    sum += __shfl_xor(sum, 16, 32);
    float inv = 1.f / sum;
    float part = 0.f;
#pragma unroll
    for (int r = 0; r < 8; ++r) {
      int m = r + 8 * grp;
      float vv = (float)v16[(size_t)(b * kN + sidx[m]) * kHid + j] + peF[m][j];
      part += vals[r] * inv * vv;
    }
    part += __shfl_xor(part, 16, 32);
    if (grp == 0) res16[(size_t)pid * kHid + j] = (f16)part;
  }
}

// ---------------- head: ag proj + residual + exact-GELU MLP + residual ------
__global__ void __launch_bounds__(32) head_kernel(const f16* __restrict__ res16,
    const float* __restrict__ feat,
    const f16* __restrict__ agwpk, const float* __restrict__ ag_b,
    const f16* __restrict__ mw1pk, const float* __restrict__ mb1,
    const f16* __restrict__ mw2pk, const float* __restrict__ mb2,
    float* __restrict__ out) {
  __shared__ alignas(16) f16 T[16][kFeat];
  __shared__ alignas(16) f16 Z[16][kHid];
  const int lane = threadIdx.x;
  const int b  = blockIdx.x / kNT;
  const int n0 = (blockIdx.x % kNT) * 16;
  const f16* Arow = res16 + (size_t)(b * kN + n0) * kHid;   // 16 contiguous rows
  const int col = lane & 15, grp = (lane >> 4) & 1;
  for (int nt = 0; nt < 4; ++nt) {
    v8f acc = {};
#pragma unroll
    for (int ks = 0; ks < 4; ++ks)
      acc = WMMA_F16(load_a(Arow, kHid, ks * 32), load_pk(agwpk, nt * 4 + ks), acc);
    int j = nt * 16 + col;
    float bb = ag_b[j];
#pragma unroll
    for (int r = 0; r < 8; ++r) {
      int row = r + 8 * grp;
      T[row][j] = (f16)(acc[r] + bb + feat[(size_t)(b * kN + n0 + row) * kFeat + j]);
    }
  }
  __syncthreads();
  for (int nt = 0; nt < 8; ++nt) {
    v8f acc = {};
#pragma unroll
    for (int ks = 0; ks < 2; ++ks)
      acc = WMMA_F16(load_a(&T[0][0], kFeat, ks * 32), load_pk(mw1pk, nt * 2 + ks), acc);
    int j = nt * 16 + col;
    float bb = mb1[j];
#pragma unroll
    for (int r = 0; r < 8; ++r) {
      float x = acc[r] + bb;
      Z[r + 8 * grp][j] = (f16)(0.5f * x * (1.f + erff(x * 0.7071067811865476f)));
    }
  }
  __syncthreads();
  for (int nt = 0; nt < 4; ++nt) {
    v8f acc = {};
#pragma unroll
    for (int ks = 0; ks < 4; ++ks)
      acc = WMMA_F16(load_a(&Z[0][0], kHid, ks * 32), load_pk(mw2pk, nt * 4 + ks), acc);
    int j = nt * 16 + col;
    float bb = mb2[j];
#pragma unroll
    for (int r = 0; r < 8; ++r) {
      int row = r + 8 * grp;
      out[(size_t)(b * kN + n0 + row) * kFeat + j] =
          acc[r] + bb + feat[(size_t)(b * kN + n0 + row) * kFeat + j];
    }
  }
}

// ---------------- launcher ----------------
extern "C" void kernel_launch(void* const* d_in, const int* in_sizes, int n_in,
                              void* d_out, int out_size, void* d_ws, size_t ws_size,
                              hipStream_t stream) {
  (void)in_sizes; (void)n_in; (void)out_size; (void)ws_size;
  const float* xyzp   = (const float*)d_in[0];
  const float* feat   = (const float*)d_in[1];
  const float* ln_g   = (const float*)d_in[2];
  const float* ln_b   = (const float*)d_in[3];
  const float* W1     = (const float*)d_in[4];
  const float* b1     = (const float*)d_in[5];
  const float* W2     = (const float*)d_in[6];
  const float* pe_w1  = (const float*)d_in[7];
  const float* pe_b1  = (const float*)d_in[8];
  const float* pe_w2  = (const float*)d_in[9];
  const float* pe_b2  = (const float*)d_in[10];
  const float* at_w1  = (const float*)d_in[11];
  const float* at_b1  = (const float*)d_in[12];
  const float* at_w2  = (const float*)d_in[13];
  const float* at_b2  = (const float*)d_in[14];
  const float* ag_w   = (const float*)d_in[15];
  const float* ag_b   = (const float*)d_in[16];
  const float* mlp_w1 = (const float*)d_in[17];
  const float* mlp_b1 = (const float*)d_in[18];
  const float* mlp_w2 = (const float*)d_in[19];
  const float* mlp_b2 = (const float*)d_in[20];

  char* ws = (char*)d_ws;
  size_t off = 0;
  auto alloc = [&](size_t bytes) -> void* {
    void* p = ws + off;
    off += (bytes + 255) & ~(size_t)255;
    return p;
  };
  int* grid_ws = (int*)alloc((size_t)kB * kGP * kGP * 4);
  int* knn_ws  = (int*)alloc((size_t)kB * kN * kK * 4);
  f16* sp16    = (f16*)alloc((size_t)kB * kN * kSPC * 2);
  f16* w1pk    = (f16*)alloc((size_t)8 * 9 * 3 * 512 * 2);   // [nt][tap][ks] frags
  f16* w2pk    = (f16*)alloc((size_t)24 * 9 * 4 * 512 * 2);
  f16* pew2pk  = (f16*)alloc((size_t)8 * 4 * 512 * 2);
  f16* atw1pk  = (f16*)alloc((size_t)8 * 4 * 512 * 2);
  f16* atw2pk  = (f16*)alloc((size_t)8 * 4 * 512 * 2);
  f16* agwpk   = (f16*)alloc((size_t)4 * 4 * 512 * 2);
  f16* mw1pk   = (f16*)alloc((size_t)8 * 2 * 512 * 2);
  f16* mw2pk   = (f16*)alloc((size_t)4 * 4 * 512 * 2);
  f16* h1_16   = (f16*)alloc((size_t)kB * kN * kHid * 2);
  f16* q16     = (f16*)alloc((size_t)kB * kN * kHid * 2);
  f16* k16     = (f16*)alloc((size_t)kB * kN * kHid * 2);
  f16* v16     = (f16*)alloc((size_t)kB * kN * kHid * 2);
  f16* res16   = (f16*)alloc((size_t)kB * kN * kHid * 2);

  auto cdiv = [](int a, int b) { return (a + b - 1) / b; };
  auto pack = [&](const float* src, f16* dst, int nmat, int ksteps, int ntiles,
                  int Nd, int kvalid) {
    int total = nmat * ksteps * ntiles * 512;
    pack_b_kernel<<<cdiv(total, 256), 256, 0, stream>>>(src, dst, nmat, ksteps, ntiles,
                                                        Nd, kvalid);
  };
  // weight pre-swizzle into WMMA B-fragment order (f16)
  pack(W1,     w1pk,   9, 3, 8,  kHid, 66);    // zero-pad 66 -> 96
  pack(W2,     w2pk,   9, 4, 24, 384,  kHid);
  pack(pe_w2,  pew2pk, 1, 4, 8,  kHid, kHid);
  pack(at_w1,  atw1pk, 1, 4, 8,  kHid, kHid);
  pack(at_w2,  atw2pk, 1, 4, 8,  kHid, kHid);
  pack(ag_w,   agwpk,  1, 4, 4,  kFeat, kHid);
  pack(mlp_w1, mw1pk,  1, 2, 8,  kHid, kFeat);
  pack(mlp_w2, mw2pk,  1, 4, 4,  kFeat, kHid);

  // grid scatter + layernorm + knn
  fill_grid_kernel<<<cdiv(kB * kGP * kGP, 256), 256, 0, stream>>>(grid_ws, kB * kGP * kGP);
  scatter_grid_kernel<<<cdiv(kB * kN, 256), 256, 0, stream>>>(xyzp, grid_ws);
  spln_kernel<<<cdiv(kB * kN * 32, 256), 256, 0, stream>>>(xyzp, feat, ln_g, ln_b, sp16);
  knn_kernel<<<kB * (kN / 256), 256, 0, stream>>>(xyzp, knn_ws);

  // conv stages (WMMA, 4 waves/block share the gathered LDS tiles)
  conv1_kernel<<<kB * kNT, 128, 0, stream>>>(xyzp, grid_ws, sp16, w1pk, b1, h1_16);
  conv2_kernel<<<kB * kNT, 128, 0, stream>>>(xyzp, grid_ws, h1_16, w2pk, q16, k16, v16);

  // attention (WMMA) + head (WMMA)
  attn_kernel<<<kB * kN, 32, 0, stream>>>(xyzp, knn_ws, q16, k16, v16,
                                          pe_w1, pe_b1, pew2pk, pe_b2,
                                          atw1pk, at_b1, atw2pk, at_b2, res16);
  head_kernel<<<kB * kNT, 32, 0, stream>>>(res16, feat, agwpk, ag_b,
                                           mw1pk, mlp_b1, mw2pk, mlp_b2, (float*)d_out);
}
